// NittaMHA_64896955843223
// MI455X (gfx1250) — compile-verified, matrix operand order checked by reading, not run on hardware
//
#include <hip/hip_runtime.h>
#include <hip/hip_bf16.h>

typedef __attribute__((ext_vector_type(16))) __bf16        v16bf;
typedef __attribute__((ext_vector_type(8)))  float         v8f;
typedef __attribute__((ext_vector_type(8)))  unsigned short ushort8;

struct bfrag { ushort8 lo, hi; };   // 32 bytes == one 16x32 bf16 WMMA operand per lane

static constexpr int Bsz = 4, S = 2048, E = 1024, H = 16, DK = 64;
static constexpr int Mtot = Bsz * S;                 // 8192
static constexpr size_t HEADSZ = (size_t)Bsz * H * S * DK;   // elements per q/k/v ws tensor

__device__ __forceinline__ unsigned short f32_to_bf16(float f) {
    unsigned int u = __builtin_bit_cast(unsigned int, f);
    unsigned int r = u + 0x7FFFu + ((u >> 16) & 1u);     // round-to-nearest-even
    return (unsigned short)(r >> 16);
}

__device__ __forceinline__ v8f wmma_bf16(const bfrag& a, const bfrag& b, v8f c) {
    return __builtin_amdgcn_wmma_f32_16x16x32_bf16(
        /*neg_a=*/false, __builtin_bit_cast(v16bf, a),
        /*neg_b=*/false, __builtin_bit_cast(v16bf, b),
        /*c_mod=*/(short)0, c, /*reuse_a=*/false, /*reuse_b=*/false);
}

// Async global -> LDS 16-byte copy (CDNA5, tracked on ASYNCcnt).
// Per ISA 10.2 the low 32 bits of a generic shared-aperture address are the
// LDS byte offset, so truncating the generic pointer gives the VDST operand.
__device__ __forceinline__ void async_copy_b128(unsigned short* lds_dst,
                                                const unsigned short* gsrc) {
    unsigned int loff = (unsigned int)(size_t)(void*)lds_dst;
    asm volatile("global_load_async_to_lds_b128 %0, %1, off"
                 :: "v"(loff), "v"((unsigned long long)(size_t)gsrc)
                 : "memory");
}
__device__ __forceinline__ void wait_async0() {
    asm volatile("s_wait_asynccnt 0x0" ::: "memory");
}

// ---------------------------------------------------------------------------
// GEMM: Out[m][n] = sum_k A[m][k] * W[n][k] + bias[n]
//   A_BF16=false : A is f32 (loaded to VGPRs, converted to bf16 into LDS)
//   A_BF16=true  : A is bf16 in workspace -> async-to-LDS copies
//   OUT_MODE 0: bf16, [B,H,S,DK]   (q / k)
//   OUT_MODE 1: bf16, [B,H,DK,S]   (v transposed for contiguous P.V reads)
//   OUT_MODE 2: f32,  row-major [M][E]  (final output)
// 256 threads = 8 waves; tile 128(M) x 128(N), K step 32, LDS double-buffered:
// next K-tile's global loads are issued before the WMMAs on the current tile.
// ---------------------------------------------------------------------------
template <bool A_BF16, int OUT_MODE>
__global__ __launch_bounds__(256) void gemm_kernel(const void* __restrict__ Aptr,
                                                   const float* __restrict__ W,
                                                   const float* __restrict__ bias,
                                                   void* __restrict__ Out) {
    __shared__ unsigned short As[2][128][40];
    __shared__ unsigned short Bs[2][128][40];

    const int tid   = threadIdx.x;
    const int m0    = blockIdx.x * 128;
    const int n0    = blockIdx.y * 128;
    const int w     = tid >> 5;
    const int lane  = tid & 31;
    const int wm    = (w & 3) * 32;
    const int wn    = (w >> 2) * 64;
    const int lhalf = lane >> 4;
    const int lmod  = lane & 15;

    float4 areg[4], wreg[4];

    auto load_w = [&](int k0) {
#pragma unroll
        for (int it = 0; it < 4; ++it) {
            int idx = (tid + it * 256) * 4;            // 0..4095
            int r = idx >> 5, c = idx & 31;
            wreg[it] = *(const float4*)&W[(size_t)(n0 + r) * E + k0 + c];
        }
    };
    auto store_w = [&](int p) {
#pragma unroll
        for (int it = 0; it < 4; ++it) {
            int idx = (tid + it * 256) * 4;
            int r = idx >> 5, c = idx & 31;
            uint2 pk;
            pk.x = (unsigned int)f32_to_bf16(wreg[it].x) |
                   ((unsigned int)f32_to_bf16(wreg[it].y) << 16);
            pk.y = (unsigned int)f32_to_bf16(wreg[it].z) |
                   ((unsigned int)f32_to_bf16(wreg[it].w) << 16);
            *(uint2*)&Bs[p][r][c] = pk;
        }
    };
    auto load_a = [&](int k0) {
        const float* A = (const float*)Aptr;
#pragma unroll
        for (int it = 0; it < 4; ++it) {
            int idx = (tid + it * 256) * 4;
            int r = idx >> 5, c = idx & 31;
            areg[it] = *(const float4*)&A[(size_t)(m0 + r) * E + k0 + c];
        }
    };
    auto store_a = [&](int p) {
#pragma unroll
        for (int it = 0; it < 4; ++it) {
            int idx = (tid + it * 256) * 4;
            int r = idx >> 5, c = idx & 31;
            uint2 pk;
            pk.x = (unsigned int)f32_to_bf16(areg[it].x) |
                   ((unsigned int)f32_to_bf16(areg[it].y) << 16);
            pk.y = (unsigned int)f32_to_bf16(areg[it].z) |
                   ((unsigned int)f32_to_bf16(areg[it].w) << 16);
            *(uint2*)&As[p][r][c] = pk;
        }
    };
    auto async_a = [&](int p, int k0) {
        const unsigned short* A = (const unsigned short*)Aptr;
#pragma unroll
        for (int it = 0; it < 2; ++it) {
            int idx = (tid + it * 256) * 8;            // 0..4095
            int r = idx >> 5, c = idx & 31;
            async_copy_b128(&As[p][r][c], &A[(size_t)(m0 + r) * E + k0 + c]);
        }
    };

    v8f acc[2][4];
#pragma unroll
    for (int i = 0; i < 2; ++i)
#pragma unroll
        for (int j = 0; j < 4; ++j) acc[i][j] = (v8f)0.0f;

    // ---- prologue: stage K-tile 0 into buffer 0 ----
    if constexpr (A_BF16) async_a(0, 0); else { load_a(0); store_a(0); }
    load_w(0);
    store_w(0);
    if constexpr (A_BF16) wait_async0();
    __syncthreads();

    for (int k0 = 0; k0 < E; k0 += 32) {
        const int p = (k0 >> 5) & 1;
        const bool next = (k0 + 32 < E);

        // issue next tile's global traffic before computing on current tile
        if (next) {
            if constexpr (A_BF16) async_a(p ^ 1, k0 + 32); else load_a(k0 + 32);
            load_w(k0 + 32);
        }

        // ---- fragments + WMMA on buffer p ----
        bfrag af[2], bf[4];
        const int kb = lhalf * 8;
#pragma unroll
        for (int ms = 0; ms < 2; ++ms) {
            af[ms].lo = *(const ushort8*)&As[p][wm + ms * 16 + lmod][kb];
            af[ms].hi = *(const ushort8*)&As[p][wm + ms * 16 + lmod][kb + 16];
        }
#pragma unroll
        for (int ns = 0; ns < 4; ++ns) {
            bf[ns].lo = *(const ushort8*)&Bs[p][wn + ns * 16 + lmod][kb];
            bf[ns].hi = *(const ushort8*)&Bs[p][wn + ns * 16 + lmod][kb + 16];
        }
#pragma unroll
        for (int ms = 0; ms < 2; ++ms)
#pragma unroll
            for (int ns = 0; ns < 4; ++ns)
                acc[ms][ns] = wmma_bf16(af[ms], bf[ns], acc[ms][ns]);

        // convert/store next tile into the alternate buffer
        if (next) {
            if constexpr (!A_BF16) store_a(p ^ 1);
            store_w(p ^ 1);
            if constexpr (A_BF16) wait_async0();
        }
        __syncthreads();
    }

    // ---- epilogue: bias + store ----
#pragma unroll
    for (int ns = 0; ns < 4; ++ns) {
        const int n  = n0 + wn + ns * 16 + lmod;
        const float bv = bias[n];
#pragma unroll
        for (int ms = 0; ms < 2; ++ms) {
#pragma unroll
            for (int r = 0; r < 8; ++r) {
                const int m   = m0 + wm + ms * 16 + lhalf * 8 + r;
                const float v = acc[ms][ns][r] + bv;
                if constexpr (OUT_MODE == 2) {
                    ((float*)Out)[(size_t)m * E + n] = v;
                } else {
                    const int b = m >> 11, s = m & (S - 1);
                    const int h = n >> 6,  d = n & (DK - 1);
                    size_t off;
                    if constexpr (OUT_MODE == 0)
                        off = (((size_t)b * H + h) * S + s) * DK + d;
                    else
                        off = (((size_t)b * H + h) * DK + d) * S + s;
                    ((unsigned short*)Out)[off] = f32_to_bf16(v);
                }
            }
        }
    }
}

// ---------------------------------------------------------------------------
// Flash attention. One block = 4 waves, same (b,h); each wave owns 16 query
// rows (64 per block). K/V tiles (64 keys) are double-buffered in LDS and
// filled with async-to-LDS copies issued a full iteration ahead, so the DMA
// overlaps the WMMA + softmax work. One barrier per iteration: the tile-ready
// barrier also serves as the WAR guard for the alternate buffer.
// Online softmax state lives in the WMMA C/D layout slots (VGPR r <-> row
// r / r+8 per 16-lane half); reductions are shfl_xor within 16-lane groups.
// ---------------------------------------------------------------------------
__global__ __launch_bounds__(128) void attn_kernel(const unsigned short* __restrict__ q,
                                                   const unsigned short* __restrict__ k,
                                                   const unsigned short* __restrict__ vT,
                                                   unsigned short* __restrict__ ctx) {
    __shared__ unsigned short Ks[2][64][72];       // [key][d]
    __shared__ unsigned short Vs[2][64][72];       // [d][key]
    __shared__ unsigned short probs[4][16][72];    // per-wave P staging, bf16

    const int tid = threadIdx.x;
    const int w = tid >> 5, lane = tid & 31;
    const int lhalf = lane >> 4, lmod = lane & 15;
    const int bh   = blockIdx.x >> 5;              // b*H + h
    const int qblk = blockIdx.x & 31;              // S/64 query super-blocks
    const size_t base = (size_t)bh * S * DK;
    const int s0 = qblk * 64 + w * 16;             // this wave's 16 query rows

    auto stage = [&](int p, int t0) {
#pragma unroll
        for (int it = 0; it < 4; ++it) {
            int e = (tid + it * 128) * 8;          // 0..4095
            int r = e >> 6, c = e & 63;
            async_copy_b128(&Ks[p][r][c], &k[base + (size_t)(t0 + r) * DK + c]);
        }
#pragma unroll
        for (int it = 0; it < 4; ++it) {
            int e = (tid + it * 128) * 8;
            int r = e >> 6, c = e & 63;
            async_copy_b128(&Vs[p][r][c], &vT[base + (size_t)r * S + t0 + c]);
        }
    };

    // q fragments: 16 rows x DK=64 -> two K32 A-operands, kept in registers
    bfrag qf[2];
#pragma unroll
    for (int kc = 0; kc < 2; ++kc) {
        const int kb = kc * 32 + lhalf * 8;
        const unsigned short* p = &q[base + (size_t)(s0 + lmod) * DK + kb];
        qf[kc].lo = *(const ushort8*)p;
        qf[kc].hi = *(const ushort8*)(p + 16);
    }

    v8f cacc[4];
#pragma unroll
    for (int dt = 0; dt < 4; ++dt) cacc[dt] = (v8f)0.0f;
    float mrun[8], srun[8];
#pragma unroll
    for (int r = 0; r < 8; ++r) { mrun[r] = -1e30f; srun[r] = 0.0f; }

    stage(0, 0);   // preload first tile

    for (int t0 = 0; t0 < S; t0 += 64) {
        const int p = (t0 >> 6) & 1;
        wait_async0();          // own async copies for tile p complete
        __syncthreads();        // whole tile present; alt buffer free to reuse
        if (t0 + 64 < S) stage(p ^ 1, t0 + 64);   // DMA next tile during compute

        // ---- scores: 16 queries x 64 keys = four 16x16 WMMA tiles over K=DK ----
        v8f sc[4];
#pragma unroll
        for (int n = 0; n < 4; ++n) {
            v8f a = (v8f)0.0f;
#pragma unroll
            for (int kc = 0; kc < 2; ++kc) {
                const int kb = kc * 32 + lhalf * 8;
                bfrag kf;
                kf.lo = *(const ushort8*)&Ks[p][n * 16 + lmod][kb];
                kf.hi = *(const ushort8*)&Ks[p][n * 16 + lmod][kb + 16];
                a = wmma_bf16(qf[kc], kf, a);
            }
            sc[n] = a * 0.125f;                    // 1/sqrt(DK)
        }

        // ---- online softmax (rows live per VGPR-slot r) ----
#pragma unroll
        for (int r = 0; r < 8; ++r) {
            float mx = fmaxf(fmaxf(sc[0][r], sc[1][r]), fmaxf(sc[2][r], sc[3][r]));
#pragma unroll
            for (int off = 1; off < 16; off <<= 1)
                mx = fmaxf(mx, __shfl_xor(mx, off, 32));
            const float mnew = fmaxf(mrun[r], mx);
            float ps = 0.0f;
#pragma unroll
            for (int n = 0; n < 4; ++n) {
                sc[n][r] = __expf(sc[n][r] - mnew);
                ps += sc[n][r];
            }
#pragma unroll
            for (int off = 1; off < 16; off <<= 1)
                ps += __shfl_xor(ps, off, 32);
            const float corr = __expf(mrun[r] - mnew);
            srun[r] = srun[r] * corr + ps;
            mrun[r] = mnew;
#pragma unroll
            for (int dt = 0; dt < 4; ++dt) cacc[dt][r] *= corr;

            // D-layout -> LDS (row = lhalf*8 + r, col = key within block)
            const int mrow = lhalf * 8 + r;
#pragma unroll
            for (int n = 0; n < 4; ++n)
                probs[w][mrow][n * 16 + lmod] = f32_to_bf16(sc[n][r]);
        }

        // ---- P (16x64) from LDS in A-layout: two K32 operands ----
        bfrag pf[2];
#pragma unroll
        for (int kc = 0; kc < 2; ++kc) {
            const int kb = kc * 32 + lhalf * 8;
            pf[kc].lo = *(const ushort8*)&probs[w][lmod][kb];
            pf[kc].hi = *(const ushort8*)&probs[w][lmod][kb + 16];
        }

        // ---- ctx += P * V  (V tile in LDS, keys contiguous per d-row) ----
#pragma unroll
        for (int dt = 0; dt < 4; ++dt) {
#pragma unroll
            for (int kc = 0; kc < 2; ++kc) {
                const int kb = kc * 32 + lhalf * 8;
                bfrag vf;
                vf.lo = *(const ushort8*)&Vs[p][dt * 16 + lmod][kb];
                vf.hi = *(const ushort8*)&Vs[p][dt * 16 + lmod][kb + 16];
                cacc[dt] = wmma_bf16(pf[kc], vf, cacc[dt]);
            }
        }
    }

    // ---- normalize + store ctx as bf16 [B, S, H*DK] (heads concatenated) ----
    const int b = bh >> 4, h = bh & 15;
#pragma unroll
    for (int dt = 0; dt < 4; ++dt) {
        const int d = h * DK + dt * 16 + lmod;
#pragma unroll
        for (int r = 0; r < 8; ++r) {
            const int srow = s0 + lhalf * 8 + r;
            ctx[((size_t)b * S + srow) * E + d] = f32_to_bf16(cacc[dt][r] / srun[r]);
        }
    }
}

// ---------------------------------------------------------------------------
extern "C" void kernel_launch(void* const* d_in, const int* in_sizes, int n_in,
                              void* d_out, int out_size, void* d_ws, size_t ws_size,
                              hipStream_t stream) {
    (void)in_sizes; (void)n_in; (void)out_size; (void)ws_size;
    const float* Q  = (const float*)d_in[0];
    const float* K  = (const float*)d_in[1];
    const float* V  = (const float*)d_in[2];
    const float* Wq = (const float*)d_in[3];
    const float* bq = (const float*)d_in[4];
    const float* Wk = (const float*)d_in[5];
    const float* bk = (const float*)d_in[6];
    const float* Wv = (const float*)d_in[7];
    const float* bv = (const float*)d_in[8];
    const float* Wo = (const float*)d_in[9];
    const float* bo = (const float*)d_in[10];

    unsigned short* qws  = (unsigned short*)d_ws;
    unsigned short* kws  = qws + HEADSZ;
    unsigned short* vws  = qws + 2 * HEADSZ;
    unsigned short* ctxw = qws + 3 * HEADSZ;   // 4 x 16 MB = 64 MB workspace

    const dim3 ggrid(Mtot / 128, E / 128);     // 64 x 8
    gemm_kernel<false, 0><<<ggrid, 256, 0, stream>>>(Q, Wq, bq, qws);
    gemm_kernel<false, 0><<<ggrid, 256, 0, stream>>>(K, Wk, bk, kws);
    gemm_kernel<false, 1><<<ggrid, 256, 0, stream>>>(V, Wv, bv, vws);  // transposed

    attn_kernel<<<Bsz * H * (S / 64), 128, 0, stream>>>(qws, kws, vws, ctxw);

    gemm_kernel<true, 2><<<ggrid, 256, 0, stream>>>(ctxw, Wo, bo, d_out);
}
